// DecoderRNN_26929444946064
// MI455X (gfx1250) — matrix operand order repaired and checked by reading, not compiled, not run on hardware
//
#include <hip/hip_runtime.h>
#include <hip/hip_bf16.h>
#include <cstddef>

// ---------------- problem constants (from reference) ----------------
#define BB   64
#define PP   196
#define ENC  2048
#define ATT  512
#define DEC  512
#define EMB  512
#define VV   10000
#define MAXLEN 30
#define TT   (MAXLEN - 1)          // 29 decode steps
#define XDIM (EMB + ENC + DEC)     // 3072 : [emb | context | h]

typedef __bf16 bf16_t;
typedef __attribute__((ext_vector_type(8)))  __bf16 v8bf;
typedef __attribute__((ext_vector_type(16))) __bf16 v16bf;
typedef __attribute__((ext_vector_type(8)))  float  v8f;

// ====================================================================
// WMMA bf16 NT GEMM:  C[M,N] = A[M,K] @ W[N,K]^T (+bias[n])
// A row-major (row stride K), W row-major (row stride K).
// One wave computes a 32x32 C macro-tile (2x2 of 16x16 WMMA tiles):
// 4 accumulators, A/B chunks loaded once per k-step and reused ->
// 1KB of operand traffic per WMMA instead of 2KB.
// Requires M%32==0, N%16==0, K%32==0. Ragged last N supertile (when
// N/16 is odd) is handled wave-uniformly: col-1 B pointer aliases col 0
// and col-1 results are not stored (EXEC stays all-ones for WMMA).
//
// CDNA5 wave32 layouts (ISA 7.12.2):
//   A 16x32 bf16 : lane l holds row M=l%16; elems 0..7  = K[hf*8 .. +7],
//                  elems 8..15 = K[16+hf*8 .. +7]        (hf = l>>4)
//   B 32x16 bf16 : lane l holds col N=l%16; elems 0..7  = K[hf*8 .. +7],
//                  elems 8..15 = K[16+hf*8 .. +7]
//   C 16x16 f32  : lane l = col N=l%16; vgpr j = row M = j + 8*hf
// ====================================================================
__global__ __launch_bounds__(128)
void gemm_bf16_nt(const bf16_t* __restrict__ A, const bf16_t* __restrict__ W,
                  const float* __restrict__ bias,
                  float* __restrict__ Cf, bf16_t* __restrict__ Cbf,
                  int M, int N, int K, int ldc)
{
    const int waveInBlock = threadIdx.x >> 5;
    const int stile       = blockIdx.x * 4 + waveInBlock;   // 32x32 supertile id
    const int ntn         = N >> 4;                         // # of 16-wide N tiles
    const int nsn         = (ntn + 1) >> 1;                 // # of 32-wide N supertiles
    const int total       = (M >> 5) * nsn;
    if (stile >= total) return;                             // wave-uniform exit

    const int sm   = stile / nsn;
    const int sn   = stile % nsn;
    const int tn0  = sn * 2;
    const bool hasN1 = (tn0 + 1) < ntn;

    const int lane = threadIdx.x & 31;
    const int r    = lane & 15;
    const int hf   = lane >> 4;

    const bf16_t* Arow0 = A + (size_t)(sm * 32 + r) * K;
    const bf16_t* Arow1 = Arow0 + (size_t)16 * K;
    const bf16_t* Wrow0 = W + (size_t)(tn0 * 16 + r) * K;
    const bf16_t* Wrow1 = hasN1 ? (Wrow0 + (size_t)16 * K) : Wrow0;

    v8f acc00 = {}, acc01 = {}, acc10 = {}, acc11 = {};

    for (int k0 = 0; k0 < K; k0 += 32) {
        v8bf a00 = *(const v8bf*)(Arow0 + k0 + hf * 8);
        v8bf a01 = *(const v8bf*)(Arow0 + k0 + 16 + hf * 8);
        v8bf a10 = *(const v8bf*)(Arow1 + k0 + hf * 8);
        v8bf a11 = *(const v8bf*)(Arow1 + k0 + 16 + hf * 8);
        v8bf b00 = *(const v8bf*)(Wrow0 + k0 + hf * 8);
        v8bf b01 = *(const v8bf*)(Wrow0 + k0 + 16 + hf * 8);
        v8bf b10 = *(const v8bf*)(Wrow1 + k0 + hf * 8);
        v8bf b11 = *(const v8bf*)(Wrow1 + k0 + 16 + hf * 8);

        v16bf av0 = __builtin_shufflevector(a00, a01, 0,1,2,3,4,5,6,7,8,9,10,11,12,13,14,15);
        v16bf av1 = __builtin_shufflevector(a10, a11, 0,1,2,3,4,5,6,7,8,9,10,11,12,13,14,15);
        v16bf bv0 = __builtin_shufflevector(b00, b01, 0,1,2,3,4,5,6,7,8,9,10,11,12,13,14,15);
        v16bf bv1 = __builtin_shufflevector(b10, b11, 0,1,2,3,4,5,6,7,8,9,10,11,12,13,14,15);

        acc00 = __builtin_amdgcn_wmma_f32_16x16x32_bf16(false, av0, false, bv0, (short)0, acc00, false, false);
        acc01 = __builtin_amdgcn_wmma_f32_16x16x32_bf16(false, av0, false, bv1, (short)0, acc01, false, false);
        acc10 = __builtin_amdgcn_wmma_f32_16x16x32_bf16(false, av1, false, bv0, (short)0, acc10, false, false);
        acc11 = __builtin_amdgcn_wmma_f32_16x16x32_bf16(false, av1, false, bv1, (short)0, acc11, false, false);
    }

    const int   col0 = tn0 * 16 + r;
    const int   col1 = col0 + 16;
    const float bb0  = bias ? bias[col0] : 0.0f;
    const float bb1  = (bias && hasN1) ? bias[col1] : 0.0f;

#pragma unroll
    for (int j = 0; j < 8; ++j) {
        const int row0 = sm * 32 + j + 8 * hf;
        const int row1 = row0 + 16;
        const float v00 = acc00[j] + bb0;
        const float v10 = acc10[j] + bb0;
        if (Cf) {
            Cf[(size_t)row0 * ldc + col0] = v00;
            Cf[(size_t)row1 * ldc + col0] = v10;
        }
        if (Cbf) {
            Cbf[(size_t)row0 * ldc + col0] = (bf16_t)v00;
            Cbf[(size_t)row1 * ldc + col0] = (bf16_t)v10;
        }
        if (hasN1) {
            const float v01 = acc01[j] + bb1;
            const float v11 = acc11[j] + bb1;
            if (Cf) {
                Cf[(size_t)row0 * ldc + col1] = v01;
                Cf[(size_t)row1 * ldc + col1] = v11;
            }
            if (Cbf) {
                Cbf[(size_t)row0 * ldc + col1] = (bf16_t)v01;
                Cbf[(size_t)row1 * ldc + col1] = (bf16_t)v11;
            }
        }
    }
}

// ---------------- helpers: conversions / packing ----------------
__global__ void k_f32_to_bf16(const float* __restrict__ s, bf16_t* __restrict__ d, size_t n)
{
    size_t i      = (size_t)blockIdx.x * blockDim.x + threadIdx.x;
    size_t stride = (size_t)gridDim.x * blockDim.x;
    for (; i < n; i += stride) d[i] = (bf16_t)s[i];
}

// W_cat[n][0:2560] = W_ih[n][:], W_cat[n][2560:3072] = W_hh[n][:]
__global__ void k_build_wcat(const float* __restrict__ Wih, const float* __restrict__ Whh,
                             bf16_t* __restrict__ Wcat)
{
    const size_t n = (size_t)(4 * DEC) * XDIM;
    size_t i       = (size_t)blockIdx.x * blockDim.x + threadIdx.x;
    size_t stride  = (size_t)gridDim.x * blockDim.x;
    for (; i < n; i += stride) {
        const size_t row = i / XDIM, j = i % XDIM;
        const float v = (j < (EMB + ENC)) ? Wih[row * (EMB + ENC) + j]
                                          : Whh[row * DEC + (j - (EMB + ENC))];
        Wcat[i] = (bf16_t)v;
    }
}

__global__ void k_add_bias(const float* __restrict__ a, const float* __restrict__ b,
                           float* __restrict__ o, int n)
{
    int i = blockIdx.x * blockDim.x + threadIdx.x;
    if (i < n) o[i] = a[i] + b[i];
}

// emb_bf[t][b][j] = bf16(emb_table[captions[b][t]][j]), t in [0,TT)
__global__ void k_embed_gather(const long long* __restrict__ captions,
                               const float* __restrict__ table, bf16_t* __restrict__ out)
{
    const int tb = blockIdx.x;           // t*BB + b
    const int t  = tb / BB, b = tb % BB;
    const long long idx = captions[(size_t)b * MAXLEN + t];
    const float* row = table + (size_t)idx * EMB;
    for (int j = threadIdx.x; j < EMB; j += blockDim.x)
        out[(size_t)tb * EMB + j] = (bf16_t)row[j];
}

// mean over P of encoder_out -> bf16 [B, ENC]
__global__ void k_meanpool(const float* __restrict__ enc, bf16_t* __restrict__ out)
{
    const int b = blockIdx.x;
    const float* e = enc + (size_t)b * PP * ENC;
    for (int j = threadIdx.x; j < ENC; j += blockDim.x) {
        float s = 0.f;
        for (int p = 0; p < PP; ++p) s += e[(size_t)p * ENC + j];
        out[(size_t)b * ENC + j] = (bf16_t)(s * (1.0f / PP));
    }
}

// ---------------- per-step fused attention kernel ----------------
// block b: e = relu(att1[b]+att2[b]) . w_full + b_full ; softmax ; context ;
// pack X[b] = [emb_t | context | h] (bf16) ; write alphas[b][t][:]
__global__ __launch_bounds__(256)
void k_attention(const float* __restrict__ att1, const float* __restrict__ att2,
                 const float* __restrict__ w_full, const float* __restrict__ b_full,
                 const float* __restrict__ enc, const bf16_t* __restrict__ emb_t,
                 const bf16_t* __restrict__ h_bf, bf16_t* __restrict__ X,
                 float* __restrict__ alphas, int t)
{
    __shared__ float s_att2[ATT];
    __shared__ float s_e[PP];
    __shared__ float s_tmp[256];

    const int b   = blockIdx.x;
    const int tid = threadIdx.x;

    for (int j = tid; j < ATT; j += 256) s_att2[j] = att2[(size_t)b * ATT + j];
    __syncthreads();

    const int wave = tid >> 5, lane = tid & 31;
    const float* a1b = att1 + (size_t)b * PP * ATT;
    for (int p = wave; p < PP; p += 8) {
        const float* row = a1b + (size_t)p * ATT;
        float sum = 0.f;
        for (int a = lane; a < ATT; a += 32) {
            float v = row[a] + s_att2[a];
            v = v > 0.f ? v : 0.f;
            sum += v * w_full[a];
        }
#pragma unroll
        for (int off = 16; off > 0; off >>= 1) sum += __shfl_xor(sum, off, 32);
        if (lane == 0) s_e[p] = sum + b_full[0];
    }
    __syncthreads();

    // softmax over PP
    float m = -1e30f;
    for (int p = tid; p < PP; p += 256) m = fmaxf(m, s_e[p]);
    s_tmp[tid] = m;
    __syncthreads();
    for (int s = 128; s > 0; s >>= 1) {
        if (tid < s) s_tmp[tid] = fmaxf(s_tmp[tid], s_tmp[tid + s]);
        __syncthreads();
    }
    m = s_tmp[0];
    __syncthreads();

    float sum = 0.f;
    for (int p = tid; p < PP; p += 256) {
        float ex = __expf(s_e[p] - m);
        s_e[p] = ex;
        sum += ex;
    }
    s_tmp[tid] = sum;
    __syncthreads();
    for (int s = 128; s > 0; s >>= 1) {
        if (tid < s) s_tmp[tid] += s_tmp[tid + s];
        __syncthreads();
    }
    const float inv = 1.0f / s_tmp[0];
    __syncthreads();

    for (int p = tid; p < PP; p += 256) {
        const float a = s_e[p] * inv;
        s_e[p] = a;
        alphas[((size_t)b * TT + t) * PP + p] = a;
    }
    __syncthreads();

    // context + pack X
    const float* eb = enc + (size_t)b * PP * ENC;
    bf16_t* Xb = X + (size_t)b * XDIM;
    for (int j = tid; j < ENC; j += 256) {
        float acc = 0.f;
        for (int p = 0; p < PP; ++p) acc += s_e[p] * eb[(size_t)p * ENC + j];
        Xb[EMB + j] = (bf16_t)acc;
    }
    for (int j = tid; j < EMB; j += 256) Xb[j] = emb_t[(size_t)b * EMB + j];
    for (int j = tid; j < DEC; j += 256) Xb[EMB + ENC + j] = h_bf[(size_t)b * DEC + j];
}

// ---------------- LSTM pointwise update ----------------
__global__ void k_lstm(const float* __restrict__ gates, float* __restrict__ c,
                       bf16_t* __restrict__ h_bf)
{
    const int idx = blockIdx.x * blockDim.x + threadIdx.x;   // 0 .. B*DEC-1
    const int b = idx / DEC, j = idx % DEC;
    const float* g = gates + (size_t)b * 4 * DEC;
    const float gi = 1.0f / (1.0f + __expf(-g[j]));
    const float gf = 1.0f / (1.0f + __expf(-g[DEC + j]));
    const float gg = tanhf(g[2 * DEC + j]);
    const float go = 1.0f / (1.0f + __expf(-g[3 * DEC + j]));
    const float cn = gf * c[idx] + gi * gg;
    c[idx]    = cn;
    h_bf[idx] = (bf16_t)(go * tanhf(cn));
}

// ====================================================================
// host launch
// ====================================================================
static inline size_t bump(size_t& off, size_t bytes)
{
    size_t cur = off;
    off += (bytes + 255) & ~(size_t)255;
    return cur;
}

static inline void launch_gemm(const bf16_t* A, const bf16_t* W, const float* bias,
                               float* Cf, bf16_t* Cbf, int M, int N, int K, int ldc,
                               hipStream_t s)
{
    const int ntn    = N / 16;
    const int nsn    = (ntn + 1) / 2;
    const int tiles  = (M / 32) * nsn;
    const int blocks = (tiles + 3) / 4;
    gemm_bf16_nt<<<blocks, 128, 0, s>>>(A, W, bias, Cf, Cbf, M, N, K, ldc);
}

extern "C" void kernel_launch(void* const* d_in, const int* in_sizes, int n_in,
                              void* d_out, int out_size, void* d_ws, size_t ws_size,
                              hipStream_t stream)
{
    (void)in_sizes; (void)n_in; (void)out_size; (void)ws_size;

    const float*      encoder_out = (const float*)d_in[0];
    const long long*  captions    = (const long long*)d_in[1];
    /* caption_lengths d_in[2] unused by reference outputs */
    const float*      emb_table   = (const float*)d_in[3];
    const float*      W_enc_att   = (const float*)d_in[4];
    const float*      b_enc_att   = (const float*)d_in[5];
    const float*      W_dec_att   = (const float*)d_in[6];
    const float*      b_dec_att   = (const float*)d_in[7];
    const float*      w_full_att  = (const float*)d_in[8];
    const float*      b_full_att  = (const float*)d_in[9];
    const float*      W_ih        = (const float*)d_in[10];
    const float*      b_ih        = (const float*)d_in[11];
    const float*      W_hh        = (const float*)d_in[12];
    const float*      b_hh        = (const float*)d_in[13];
    const float*      W_init_h    = (const float*)d_in[14];
    const float*      b_init_h    = (const float*)d_in[15];
    const float*      W_init_c    = (const float*)d_in[16];
    const float*      b_init_c    = (const float*)d_in[17];
    const float*      W_fc        = (const float*)d_in[18];
    const float*      b_fc        = (const float*)d_in[19];

    float* pred   = (float*)d_out;                                // [B, T, V]
    float* alphas = (float*)d_out + (size_t)BB * TT * VV;         // [B, T, P]

    // ---- workspace layout ----
    char*  ws  = (char*)d_ws;
    size_t off = 0;
    bf16_t* enc_bf     = (bf16_t*)(ws + bump(off, (size_t)BB * PP * ENC * 2));
    float*  att1       = (float*) (ws + bump(off, (size_t)BB * PP * ATT * 4));
    bf16_t* Wenc_bf    = (bf16_t*)(ws + bump(off, (size_t)ATT * ENC * 2));
    bf16_t* Wdec_bf    = (bf16_t*)(ws + bump(off, (size_t)ATT * DEC * 2));
    bf16_t* Winith_bf  = (bf16_t*)(ws + bump(off, (size_t)DEC * ENC * 2));
    bf16_t* Winitc_bf  = (bf16_t*)(ws + bump(off, (size_t)DEC * ENC * 2));
    bf16_t* Wfc_bf     = (bf16_t*)(ws + bump(off, (size_t)VV * DEC * 2));
    bf16_t* Wcat_bf    = (bf16_t*)(ws + bump(off, (size_t)(4 * DEC) * XDIM * 2));
    float*  bgates     = (float*) (ws + bump(off, (size_t)(4 * DEC) * 4));
    bf16_t* emb_bf     = (bf16_t*)(ws + bump(off, (size_t)TT * BB * EMB * 2));
    bf16_t* meanenc_bf = (bf16_t*)(ws + bump(off, (size_t)BB * ENC * 2));
    bf16_t* h_bf       = (bf16_t*)(ws + bump(off, (size_t)BB * DEC * 2));
    float*  c_f32      = (float*) (ws + bump(off, (size_t)BB * DEC * 4));
    float*  att2       = (float*) (ws + bump(off, (size_t)BB * ATT * 4));
    float*  gates      = (float*) (ws + bump(off, (size_t)BB * 4 * DEC * 4));
    bf16_t* X_bf       = (bf16_t*)(ws + bump(off, (size_t)BB * XDIM * 2));

    // ---- one-time prep ----
    k_f32_to_bf16<<<2048, 256, 0, stream>>>(encoder_out, enc_bf, (size_t)BB * PP * ENC);
    k_f32_to_bf16<<<512,  256, 0, stream>>>(W_enc_att, Wenc_bf, (size_t)ATT * ENC);
    k_f32_to_bf16<<<256,  256, 0, stream>>>(W_dec_att, Wdec_bf, (size_t)ATT * DEC);
    k_f32_to_bf16<<<512,  256, 0, stream>>>(W_init_h, Winith_bf, (size_t)DEC * ENC);
    k_f32_to_bf16<<<512,  256, 0, stream>>>(W_init_c, Winitc_bf, (size_t)DEC * ENC);
    k_f32_to_bf16<<<1024, 256, 0, stream>>>(W_fc, Wfc_bf, (size_t)VV * DEC);
    k_build_wcat<<<2048, 256, 0, stream>>>(W_ih, W_hh, Wcat_bf);
    k_add_bias<<<8, 256, 0, stream>>>(b_ih, b_hh, bgates, 4 * DEC);
    k_embed_gather<<<TT * BB, 128, 0, stream>>>(captions, emb_table, emb_bf);
    k_meanpool<<<BB, 256, 0, stream>>>(encoder_out, meanenc_bf);

    // h0 (bf16), c0 (f32)
    launch_gemm(meanenc_bf, Winith_bf, b_init_h, nullptr, h_bf, BB, DEC, ENC, DEC, stream);
    launch_gemm(meanenc_bf, Winitc_bf, b_init_c, c_f32, nullptr, BB, DEC, ENC, DEC, stream);

    // hoisted: att1 = encoder_out @ W_enc_att.T + b   [B*P, ATT]
    launch_gemm(enc_bf, Wenc_bf, b_enc_att, att1, nullptr, BB * PP, ATT, ENC, ATT, stream);

    // ---- 29 sequential decode steps ----
    for (int t = 0; t < TT; ++t) {
        // att2 = h @ W_dec_att.T + b
        launch_gemm(h_bf, Wdec_bf, b_dec_att, att2, nullptr, BB, ATT, DEC, ATT, stream);

        // attention + softmax + context + pack X + write alphas
        k_attention<<<BB, 256, 0, stream>>>(att1, att2, w_full_att, b_full_att,
                                            encoder_out, emb_bf + (size_t)t * BB * EMB,
                                            h_bf, X_bf, alphas, t);

        // gates = [emb|ctx|h] @ [W_ih|W_hh].T + (b_ih + b_hh)
        launch_gemm(X_bf, Wcat_bf, bgates, gates, nullptr, BB, 4 * DEC, XDIM, 4 * DEC, stream);

        // pointwise LSTM: c, h (h kept as bf16 for downstream GEMMs)
        k_lstm<<<(BB * DEC) / 256, 256, 0, stream>>>(gates, c_f32, h_bf);

        // preds[:, t, :] = h @ W_fc.T + b_fc   (strided directly into d_out)
        launch_gemm(h_bf, Wfc_bf, b_fc, pred + (size_t)t * VV, nullptr,
                    BB, VV, DEC, TT * VV, stream);
    }
}
